// PointnetSAModuleMSG_WithSampling_45835890983490
// MI455X (gfx1250) — matrix operand order repaired and checked by reading, not compiled, not run on hardware
//
#include <hip/hip_runtime.h>

// ---------------------------------------------------------------------------
// PointNet++ MSG SA module for MI455X (gfx1250, wave32, WMMA bf16 16x16x32)
// ---------------------------------------------------------------------------

typedef __attribute__((ext_vector_type(16))) __bf16 v16bf;
typedef __attribute__((ext_vector_type(8)))  float  v8f;

#define B_   8
#define N_   16384
#define CIN  64
#define M_   1024
#define NS1  16
#define NS2  32
#define R1SQ (0.8f * 0.8f)
#define R2SQ (1.6f * 1.6f)
#define EPS_ 1e-5f

__device__ __forceinline__ unsigned short f2bf(float x) {
  unsigned u = __float_as_uint(x);
  unsigned r = u + 0x7FFFu + ((u >> 16) & 1u);   // round-to-nearest-even
  return (unsigned short)(r >> 16);
}

// ------------------------- FPS: one workgroup per batch ---------------------
// 1024 threads, each owns 16 points' running min-distances in registers.
// xyz (192KB/batch) stays L2-resident across the 1023 iterations.
__global__ void fps_kernel(const float* __restrict__ xyz, int* __restrict__ fps_idx) {
  __shared__ float red_v[1024];
  __shared__ int   red_i[1024];
  __shared__ float cur[3];
  const int b = blockIdx.x, t = threadIdx.x;
  const float* xb = xyz + (size_t)b * N_ * 3;
  float d[16];
#pragma unroll
  for (int i = 0; i < 16; ++i) d[i] = 1e10f;
  int last = 0;
  if (t == 0) fps_idx[b * M_] = 0;
  for (int it = 1; it < M_; ++it) {
    if (t == 0) { cur[0] = xb[last * 3]; cur[1] = xb[last * 3 + 1]; cur[2] = xb[last * 3 + 2]; }
    __syncthreads();
    const float cx = cur[0], cy = cur[1], cz = cur[2];
    float best = -1.0f; int bi = 0;
#pragma unroll
    for (int i = 0; i < 16; ++i) {
      const int p = i * 1024 + t;
      const float dx = xb[p * 3] - cx, dy = xb[p * 3 + 1] - cy, dz = xb[p * 3 + 2] - cz;
      const float dd = dx * dx + dy * dy + dz * dz;
      d[i] = fminf(d[i], dd);
      if (d[i] > best) { best = d[i]; bi = p; }
    }
    red_v[t] = best; red_i[t] = bi;
    __syncthreads();
    for (int s = 512; s > 0; s >>= 1) {
      if (t < s && red_v[t + s] > red_v[t]) { red_v[t] = red_v[t + s]; red_i[t] = red_i[t + s]; }
      __syncthreads();
    }
    last = red_i[0];
    if (t == 0) fps_idx[b * M_ + it] = last;
    __syncthreads();
  }
}

// ------------------------- gather new_xyz ----------------------------------
__global__ void gather_new_xyz_kernel(const float* __restrict__ xyz,
                                      const int* __restrict__ fps_idx,
                                      float* __restrict__ new_xyz) {
  const int g = blockIdx.x * blockDim.x + threadIdx.x;
  if (g >= B_ * M_) return;
  const int b = g / M_;
  const int src = fps_idx[g];
  const float* p = xyz + ((size_t)b * N_ + src) * 3;
  new_xyz[g * 3 + 0] = p[0]; new_xyz[g * 3 + 1] = p[1]; new_xyz[g * 3 + 2] = p[2];
}

// ------------------------- ball query: one wave32 per query, both radii ----
__global__ void ball_query_kernel(const float* __restrict__ xyz,
                                  const float* __restrict__ new_xyz,
                                  int* __restrict__ idx1, int* __restrict__ idx2) {
  const int g = blockIdx.x * (blockDim.x >> 5) + (threadIdx.x >> 5);
  const int lane = threadIdx.x & 31;
  if (g >= B_ * M_) return;
  const int b = g / M_;
  const float qx = new_xyz[g * 3], qy = new_xyz[g * 3 + 1], qz = new_xyz[g * 3 + 2];
  const float* xb = xyz + (size_t)b * N_ * 3;
  int c1 = 0, c2 = 0, f1 = 0, f2 = 0;
  const unsigned lt = (1u << lane) - 1u;
  for (int base = 0; base < N_; base += 32) {
    const int p = base + lane;
    const float dx = xb[p * 3] - qx, dy = xb[p * 3 + 1] - qy, dz = xb[p * 3 + 2] - qz;
    const float d2 = dx * dx + dy * dy + dz * dz;
    const unsigned m1 = __builtin_amdgcn_ballot_w32(d2 < R1SQ);
    const unsigned m2 = __builtin_amdgcn_ballot_w32(d2 < R2SQ);
    if (d2 < R1SQ) { const int pos = c1 + __popc(m1 & lt); if (pos < NS1) idx1[(size_t)g * NS1 + pos] = p; }
    if (d2 < R2SQ) { const int pos = c2 + __popc(m2 & lt); if (pos < NS2) idx2[(size_t)g * NS2 + pos] = p; }
    if (c1 == 0 && m1) f1 = base + __builtin_ctz(m1);
    if (c2 == 0 && m2) f2 = base + __builtin_ctz(m2);
    c1 += __popc(m1); c2 += __popc(m2);
    if (c1 >= NS1 && c2 >= NS2) break;
  }
  const int fill1 = (c1 > 0) ? f1 : 0;
  const int fill2 = (c2 > 0) ? f2 : 0;
  for (int s = lane; s < NS1; s += 32) if (s >= c1) idx1[(size_t)g * NS1 + s] = fill1;
  for (int s = lane; s < NS2; s += 32) if (s >= c2) idx2[(size_t)g * NS2 + s] = fill2;
}

// ------------------------- grouping: build bf16 A [rows, 96] ---------------
// channels 0..2 = xyz - center, 3..66 = features, 67..95 = zero pad
__global__ void group_kernel(const float* __restrict__ xyz,
                             const float* __restrict__ feat,
                             const float* __restrict__ new_xyz,
                             const int* __restrict__ idx,
                             unsigned short* __restrict__ A, int S, int total) {
  const int row = blockIdx.x * blockDim.x + threadIdx.x;
  if (row >= total) return;
  const int s = row % S, bm = row / S, b = bm / M_;
  const int p = idx[(size_t)bm * S + s];
  const float* q = new_xyz + (size_t)bm * 3;
  const float* pt = xyz + ((size_t)b * N_ + p) * 3;
  unsigned short* out = A + (size_t)row * 96;
  out[0] = f2bf(pt[0] - q[0]); out[1] = f2bf(pt[1] - q[1]); out[2] = f2bf(pt[2] - q[2]);
  const float* f = feat + (size_t)b * CIN * N_ + p;
#pragma unroll 8
  for (int c = 0; c < CIN; ++c) out[3 + c] = f2bf(f[(size_t)c * N_]);
  for (int c = 3 + CIN; c < 96; ++c) out[c] = 0;
}

// ------------------------- weight convert + pad to bf16 --------------------
__global__ void cvt_weight_kernel(const float* __restrict__ w, unsigned short* __restrict__ out,
                                  int cout, int kin, int kpad) {
  const int i = blockIdx.x * blockDim.x + threadIdx.x;
  if (i >= cout * kpad) return;
  const int o = i / kpad, k = i % kpad;
  out[i] = (k < kin) ? f2bf(w[(size_t)o * kin + k]) : (unsigned short)0;
}

__global__ void zero_kernel(float* p, int n) {
  const int i = blockIdx.x * blockDim.x + threadIdx.x;
  if (i < n) p[i] = 0.0f;
}

// ------------------------- WMMA bf16 GEMM: C = A * W^T + BN stats ----------
// A: [rows, K] bf16 row-major. W: [Cout, K] bf16 row-major. 4 waves/block,
// each wave computes one 16x16 tile via v_wmma_f32_16x16x32_bf16.
union FragBF { v16bf v; unsigned u[8]; };

__global__ void gemm_bf16_kernel(const unsigned short* __restrict__ A,
                                 const unsigned short* __restrict__ W,
                                 float* __restrict__ C,
                                 float* __restrict__ sum, float* __restrict__ sumsq,
                                 int rows, int K, int Cout) {
  const int wave = threadIdx.x >> 5;
  const int lane = threadIdx.x & 31;
  const int mtile = blockIdx.x * 4 + wave;
  const int ntile = blockIdx.y;
  const int row = mtile * 16 + (lane & 15);
  const int col = ntile * 16 + (lane & 15);
  const int kbA = (lane < 16) ? 0 : 8;    // CDNA5 16-bit A-matrix lane layout
  const int kbB = (lane < 16) ? 0 : 8;    // CDNA5 16-bit B-matrix lane layout
  const unsigned short* arow = A + (size_t)row * K;
  const unsigned short* wrow = W + (size_t)col * K + kbB;
  v8f acc = {};
  for (int kc = 0; kc < K; kc += 32) {
    __builtin_prefetch(arow + kc + 32, 0, 1);
    FragBF a, bf;
#pragma unroll
    for (int j = 0; j < 8; ++j) {
      const int k = kbA + ((j < 4) ? 2 * j : 8 + 2 * j);   // VGPR j -> K pairs
      a.u[j] = *(const unsigned*)(arow + kc + k);
    }
#pragma unroll
    for (int j = 0; j < 8; ++j)
      bf.u[j] = *(const unsigned*)(wrow + kc + ((j < 4) ? 2 * j : 8 + 2 * j));
    acc = __builtin_amdgcn_wmma_f32_16x16x32_bf16(false, a.v, false, bf.v,
                                                  (short)0, acc, false, false);
  }
  const int rbase = mtile * 16 + ((lane < 16) ? 0 : 8);
  float s = 0.0f, s2 = 0.0f;
#pragma unroll
  for (int r = 0; r < 8; ++r) {
    const float v = acc[r];
    C[(size_t)(rbase + r) * Cout + col] = v;
    s += v; s2 += v * v;
  }
  atomicAdd(&sum[col], s);
  atomicAdd(&sumsq[col], s2);
}

// ------------------------- BN finalize: fold into affine -------------------
__global__ void bn_finalize_kernel(const float* __restrict__ sum, const float* __restrict__ sumsq,
                                   const float* __restrict__ g, const float* __restrict__ bta,
                                   float* __restrict__ scale, float* __restrict__ shift,
                                   int C, float inv_cnt) {
  const int c = threadIdx.x;
  if (c >= C) return;
  const float m = sum[c] * inv_cnt;
  const float v = fmaxf(sumsq[c] * inv_cnt - m * m, 0.0f);
  const float a = g[c] * rsqrtf(v + EPS_);
  scale[c] = a;
  shift[c] = bta[c] - m * a;
}

// ------------------------- BN + ReLU + cvt to bf16 (next layer input) ------
__global__ void bnrelu_cvt_kernel(const float* __restrict__ x,
                                  const float* __restrict__ scale, const float* __restrict__ shift,
                                  unsigned short* __restrict__ out, int total, int C) {
  const int i = blockIdx.x * blockDim.x + threadIdx.x;
  if (i >= total) return;
  const int c = i % C;
  out[i] = f2bf(fmaxf(fmaf(x[i], scale[c], shift[c]), 0.0f));
}

// ------------------------- BN + ReLU + max over S, pack into Aagg ----------
__global__ void maxpool_kernel(const float* __restrict__ x,  // [BM*S, 128]
                               const float* __restrict__ scale, const float* __restrict__ shift,
                               unsigned short* __restrict__ Aagg, int S, int coff) {
  const int i = blockIdx.x * blockDim.x + threadIdx.x;
  if (i >= B_ * M_ * 128) return;
  const int c = i % 128, bm = i / 128;
  const float a = scale[c], sh = shift[c];
  const float* p = x + (size_t)bm * S * 128 + c;
  float mx = -1e30f;
  for (int s = 0; s < S; ++s) mx = fmaxf(mx, fmaf(p[(size_t)s * 128], a, sh));
  mx = fmaxf(mx, 0.0f);                         // relu(max) == max(relu)
  Aagg[(size_t)bm * 256 + coff + c] = f2bf(mx);
}

// ------------------------- agg BN+ReLU -> output x (transposed) + Aconf ----
__global__ void agg_out_kernel(const float* __restrict__ tmpA,
                               const float* __restrict__ scale, const float* __restrict__ shift,
                               float* __restrict__ out_x, unsigned short* __restrict__ Aconf) {
  const int i = blockIdx.x * blockDim.x + threadIdx.x;
  if (i >= B_ * M_ * 256) return;
  const int c = i % 256, bm = i / 256, b = bm / M_, m = bm % M_;
  const float v = fmaxf(fmaf(tmpA[i], scale[c], shift[c]), 0.0f);
  out_x[((size_t)b * 256 + c) * M_ + m] = v;
  Aconf[i] = f2bf(v);
}

// ------------------------- conf BN+ReLU + head dot -------------------------
__global__ void head_kernel(const float* __restrict__ tmpC,
                            const float* __restrict__ scale, const float* __restrict__ shift,
                            const float* __restrict__ hw, const float* __restrict__ hb,
                            float* __restrict__ cls) {
  const int bm = blockIdx.x * blockDim.x + threadIdx.x;
  if (bm >= B_ * M_) return;
  const float* p = tmpC + (size_t)bm * 128;
  float acc = hb[0];
#pragma unroll 8
  for (int c = 0; c < 128; ++c) {
    const float h = fmaxf(fmaf(p[c], scale[c], shift[c]), 0.0f);
    acc = fmaf(h, hw[c], acc);
  }
  cls[bm] = acc;
}

// ---------------------------------------------------------------------------
extern "C" void kernel_launch(void* const* d_in, const int* in_sizes, int n_in,
                              void* d_out, int out_size, void* d_ws, size_t ws_size,
                              hipStream_t stream) {
  (void)n_in; (void)out_size; (void)ws_size;
  const float* xyz  = (const float*)d_in[0];
  const float* feat = (const float*)d_in[1];

  // params flattening order: detect insertion-order (mlps first, w=4288) vs
  // sorted-key order (agg first, w=65536).
  const bool sorted = (in_sizes[2] == 65536);
  int wi[6], hwI, hbI;   // conv order: m00, m01, m10, m11, agg, conf
  if (!sorted) { wi[0]=2; wi[1]=5; wi[2]=8; wi[3]=11; wi[4]=14; wi[5]=17; hwI=20; hbI=21; }
  else         { wi[4]=2; wi[5]=5; hwI=8;  hbI=9;  wi[0]=10; wi[1]=13; wi[2]=16; wi[3]=19; }
  const float *cw[6], *cg[6], *cb[6];
  for (int i = 0; i < 6; ++i) { cw[i]=(const float*)d_in[wi[i]]; cg[i]=(const float*)d_in[wi[i]+1]; cb[i]=(const float*)d_in[wi[i]+2]; }
  const float* hw = (const float*)d_in[hwI];
  const float* hb = (const float*)d_in[hbI];

  // workspace layout
  char* base = (char*)d_ws;
  size_t off = 0;
  auto ALLOC = [&](size_t bytes) { size_t r = off; off += (bytes + 255) & ~(size_t)255; return r; };
  const size_t o_fps   = ALLOC((size_t)B_ * M_ * 4);
  const size_t o_idx1  = ALLOC((size_t)B_ * M_ * NS1 * 4);
  const size_t o_idx2  = ALLOC((size_t)B_ * M_ * NS2 * 4);
  const size_t o_sum   = ALLOC(256 * 4);
  const size_t o_sumsq = ALLOC(256 * 4);
  const size_t o_scale = ALLOC(256 * 4);
  const size_t o_shift = ALLOC(256 * 4);
  const size_t o_wb[6] = { ALLOC(64*96*2), ALLOC(128*64*2), ALLOC(96*96*2),
                           ALLOC(128*96*2), ALLOC(256*256*2), ALLOC(128*256*2) };
  const size_t o_aagg  = ALLOC((size_t)B_ * M_ * 256 * 2);
  const size_t o_aconf = ALLOC((size_t)B_ * M_ * 256 * 2);
  const size_t o_A     = ALLOC((size_t)262144 * 96 * 2);
  const size_t o_Ab    = ALLOC((size_t)262144 * 96 * 2);
  const size_t o_t0    = ALLOC((size_t)262144 * 96 * 4);
  const size_t o_t1    = ALLOC((size_t)262144 * 128 * 4);

  int*   fps_idx = (int*)(base + o_fps);
  int*   idx1    = (int*)(base + o_idx1);
  int*   idx2    = (int*)(base + o_idx2);
  float* sum     = (float*)(base + o_sum);
  float* sumsq   = (float*)(base + o_sumsq);
  float* scale   = (float*)(base + o_scale);
  float* shift   = (float*)(base + o_shift);
  unsigned short* Aagg  = (unsigned short*)(base + o_aagg);
  unsigned short* Aconf = (unsigned short*)(base + o_aconf);
  unsigned short* Abuf  = (unsigned short*)(base + o_A);
  unsigned short* Ab    = (unsigned short*)(base + o_Ab);
  float* t0 = (float*)(base + o_t0);
  float* t1 = (float*)(base + o_t1);

  float* out_f    = (float*)d_out;
  float* new_xyz  = out_f;                                  // [B, M, 3]
  float* out_x    = out_f + (size_t)B_ * M_ * 3;            // [B, 256, M]
  float* out_cls  = out_x + (size_t)B_ * 256 * M_;          // [B, M, 1]

  // --- sampling / grouping ---
  fps_kernel<<<B_, 1024, 0, stream>>>(xyz, fps_idx);
  gather_new_xyz_kernel<<<(B_ * M_ + 255) / 256, 256, 0, stream>>>(xyz, fps_idx, new_xyz);
  ball_query_kernel<<<(B_ * M_) / 8, 256, 0, stream>>>(xyz, new_xyz, idx1, idx2);

  // --- weight conversion (bf16, K padded: 67->96) ---
  const int wcout[6] = { 64, 128, 96, 128, 256, 128 };
  const int wkin [6] = { 67, 64,  67, 96,  256, 256 };
  const int wkpad[6] = { 96, 64,  96, 96,  256, 256 };
  for (int i = 0; i < 6; ++i) {
    const int n = wcout[i] * wkpad[i];
    cvt_weight_kernel<<<(n + 255) / 256, 256, 0, stream>>>(cw[i], (unsigned short*)(base + o_wb[i]),
                                                           wcout[i], wkin[i], wkpad[i]);
  }

  // --- two MSG branches ---
  for (int r = 0; r < 2; ++r) {
    const int S    = r ? NS2 : NS1;
    const int C1   = r ? 96  : 64;
    const int rows = B_ * M_ * S;
    const int* idx = r ? idx2 : idx1;
    const unsigned short* W0 = (unsigned short*)(base + o_wb[2 * r]);
    const unsigned short* W1 = (unsigned short*)(base + o_wb[2 * r + 1]);
    const float *g0 = cg[2*r], *b0 = cb[2*r], *g1 = cg[2*r+1], *b1 = cb[2*r+1];

    group_kernel<<<rows / 256, 256, 0, stream>>>(xyz, feat, new_xyz, idx, Abuf, S, rows);

    // layer 0: [rows,96] x [C1,96]^T
    zero_kernel<<<2, 256, 0, stream>>>(sum, 512);
    gemm_bf16_kernel<<<dim3(rows / 64, C1 / 16), 128, 0, stream>>>(Abuf, W0, t0, sum, sumsq, rows, 96, C1);
    bn_finalize_kernel<<<1, 256, 0, stream>>>(sum, sumsq, g0, b0, scale, shift, C1, 1.0f / rows);
    bnrelu_cvt_kernel<<<((size_t)rows * C1 + 255) / 256, 256, 0, stream>>>(t0, scale, shift, Ab, rows * C1, C1);

    // layer 1: [rows,C1] x [128,C1]^T
    zero_kernel<<<2, 256, 0, stream>>>(sum, 512);
    gemm_bf16_kernel<<<dim3(rows / 64, 128 / 16), 128, 0, stream>>>(Ab, W1, t1, sum, sumsq, rows, C1, 128);
    bn_finalize_kernel<<<1, 256, 0, stream>>>(sum, sumsq, g1, b1, scale, shift, 128, 1.0f / rows);

    // BN + ReLU + max over S -> Aagg column block [.. , r*128 .. r*128+127]
    maxpool_kernel<<<(B_ * M_ * 128) / 256, 256, 0, stream>>>(t1, scale, shift, Aagg, S, r * 128);
  }

  // --- aggregation: [8192,256] x [256,256]^T ---
  const int BM = B_ * M_;
  zero_kernel<<<2, 256, 0, stream>>>(sum, 512);
  gemm_bf16_kernel<<<dim3(BM / 64, 256 / 16), 128, 0, stream>>>(Aagg, (unsigned short*)(base + o_wb[4]),
                                                                t0, sum, sumsq, BM, 256, 256);
  bn_finalize_kernel<<<1, 256, 0, stream>>>(sum, sumsq, cg[4], cb[4], scale, shift, 256, 1.0f / BM);
  agg_out_kernel<<<(BM * 256) / 256, 256, 0, stream>>>(t0, scale, shift, out_x, Aconf);

  // --- confidence: [8192,256] x [128,256]^T ---
  zero_kernel<<<2, 256, 0, stream>>>(sum, 512);
  gemm_bf16_kernel<<<dim3(BM / 64, 128 / 16), 128, 0, stream>>>(Aconf, (unsigned short*)(base + o_wb[5]),
                                                                t1, sum, sumsq, BM, 256, 128);
  bn_finalize_kernel<<<1, 256, 0, stream>>>(sum, sumsq, cg[5], cb[5], scale, shift, 128, 1.0f / BM);

  // --- head ---
  head_kernel<<<BM / 256, 256, 0, stream>>>(t1, scale, shift, hw, hb, out_cls);
}
